// GIN_62130996904043
// MI455X (gfx1250) — matrix operand order repaired and checked by reading, not compiled, not run on hardware
//
#include <hip/hip_runtime.h>
#include <hip/hip_bf16.h>

typedef __attribute__((ext_vector_type(16))) _Float16 v16h;
typedef __attribute__((ext_vector_type(8)))  float    v8f;

#define GIN_N_NODES 50000
#define GIN_N_EDGES 800000

// ---------------------------------------------------------------- utilities
__global__ void gin_zero_f32(float* __restrict__ p, int n) {
  int i = blockIdx.x * blockDim.x + threadIdx.x;
  if (i < n) p[i] = 0.0f;
}

// one thread per (edge, channel); wave = 32 contiguous channels of one edge
template<int LOGC>
__global__ void gin_scatter_add(const float* __restrict__ feat,
                                const long long* __restrict__ src,
                                const long long* __restrict__ dst,
                                float* __restrict__ agg) {
  const int C = 1 << LOGC;
  int t = blockIdx.x * blockDim.x + threadIdx.x;
  if (t >= (GIN_N_EDGES << LOGC)) return;
  int e = t >> LOGC;
  int c = t & (C - 1);
  long long s = src[e];
  long long d = dst[e];
  float v = feat[(size_t)s * C + c];
  atomicAdd(&agg[(size_t)d * C + c], v);
}

// ---------------------------------------------------------------- WMMA GEMM
// out[M x Nout] = act( in @ W[K x Nw] + bias ),  Nw <= NTILES*16
// in = COMBINE ? (1+eps)*A0 + A1 : A0
// Block: 128 threads = 4 waves; each wave owns a 16-row M slice and all N tiles.
template<int K, int NTILES, bool COMBINE>
__global__ void __launch_bounds__(128)
gin_gemm_wmma(const float* __restrict__ A0, const float* __restrict__ A1,
              const float* __restrict__ epsPtr,
              const float* __restrict__ W, const float* __restrict__ bias,
              int Nw, float* __restrict__ out, int ldo, int Nout,
              int M, int relu) {
  constexpr int NCHUNK = K / 32;
  // W pre-swizzled into per-wave fragment layout:
  // Bsf[((chunk*NTILES + tile)*32 + lane)*16 + elem], elem -> k = 32*chunk + (elem&7) + 16*(elem>=8) + 8*(lane>=16)
  __shared__ alignas(32) _Float16 Bsf[NCHUNK * NTILES * 32 * 16];
  __shared__ float bias_s[NTILES * 16];

  const int tid = threadIdx.x;
  for (int idx = tid; idx < NCHUNK * NTILES * 512; idx += blockDim.x) {
    int i    = idx & 15;
    int ln   = (idx >> 4) & 31;
    int tt   = (idx >> 9) % NTILES;
    int ch   = idx / (512 * NTILES);
    int k    = 32 * ch + (i & 7) + ((i >> 3) << 4) + ((ln >= 16) ? 8 : 0);
    int n    = 16 * tt + (ln & 15);
    Bsf[idx] = (_Float16)((n < Nw) ? W[(size_t)k * Nw + n] : 0.0f);
  }
  for (int idx = tid; idx < NTILES * 16; idx += blockDim.x)
    bias_s[idx] = (idx < Nw) ? bias[idx] : 0.0f;
  __syncthreads();

  const int wave = tid >> 5;
  const int lane = tid & 31;
  const int m0   = blockIdx.x * 64 + wave * 16;
  const int mrow = m0 + (lane & 15);
  const int koff = (lane >= 16) ? 8 : 0;   // A-layout lane split (ISA 16-bit A 16x32)
  const float epsp1 = COMBINE ? (1.0f + epsPtr[0]) : 1.0f;

  v8f acc[NTILES];
  #pragma unroll
  for (int t = 0; t < NTILES; ++t)
    #pragma unroll
    for (int j = 0; j < 8; ++j) acc[t][j] = 0.0f;

  for (int c = 0; c < NCHUNK; ++c) {
    const int kk = c * 32;
    v16h afrag;
    if (mrow < M) {
      // elems 0..7 -> k = kk+koff+0..7 ; elems 8..15 -> k = kk+16+koff+0..7
      const float* base0 = A0 + (size_t)mrow * K + kk + koff;
      float4 x0 = *(const float4*)(base0);
      float4 x1 = *(const float4*)(base0 + 4);
      float4 x2 = *(const float4*)(base0 + 16);
      float4 x3 = *(const float4*)(base0 + 20);
      if (COMBINE) {
        const float* base1 = A1 + (size_t)mrow * K + kk + koff;
        float4 g0 = *(const float4*)(base1);
        float4 g1 = *(const float4*)(base1 + 4);
        float4 g2 = *(const float4*)(base1 + 16);
        float4 g3 = *(const float4*)(base1 + 20);
        x0.x = fmaf(epsp1, x0.x, g0.x); x0.y = fmaf(epsp1, x0.y, g0.y);
        x0.z = fmaf(epsp1, x0.z, g0.z); x0.w = fmaf(epsp1, x0.w, g0.w);
        x1.x = fmaf(epsp1, x1.x, g1.x); x1.y = fmaf(epsp1, x1.y, g1.y);
        x1.z = fmaf(epsp1, x1.z, g1.z); x1.w = fmaf(epsp1, x1.w, g1.w);
        x2.x = fmaf(epsp1, x2.x, g2.x); x2.y = fmaf(epsp1, x2.y, g2.y);
        x2.z = fmaf(epsp1, x2.z, g2.z); x2.w = fmaf(epsp1, x2.w, g2.w);
        x3.x = fmaf(epsp1, x3.x, g3.x); x3.y = fmaf(epsp1, x3.y, g3.y);
        x3.z = fmaf(epsp1, x3.z, g3.z); x3.w = fmaf(epsp1, x3.w, g3.w);
      }
      afrag[0]  = (_Float16)x0.x; afrag[1]  = (_Float16)x0.y;
      afrag[2]  = (_Float16)x0.z; afrag[3]  = (_Float16)x0.w;
      afrag[4]  = (_Float16)x1.x; afrag[5]  = (_Float16)x1.y;
      afrag[6]  = (_Float16)x1.z; afrag[7]  = (_Float16)x1.w;
      afrag[8]  = (_Float16)x2.x; afrag[9]  = (_Float16)x2.y;
      afrag[10] = (_Float16)x2.z; afrag[11] = (_Float16)x2.w;
      afrag[12] = (_Float16)x3.x; afrag[13] = (_Float16)x3.y;
      afrag[14] = (_Float16)x3.z; afrag[15] = (_Float16)x3.w;
    } else {
      #pragma unroll
      for (int i = 0; i < 16; ++i) afrag[i] = (_Float16)0.0f;
    }
    #pragma unroll
    for (int t = 0; t < NTILES; ++t) {
      const v16h bfrag = *reinterpret_cast<const v16h*>(
          &Bsf[((c * NTILES + t) * 32 + lane) * 16]);
      acc[t] = __builtin_amdgcn_wmma_f32_16x16x32_f16(
          false, afrag, false, bfrag, (short)0, acc[t], false, false);
    }
  }

  // C/D layout: elem j -> m = m0 + j + 8*(lane>=16), n = tile*16 + (lane&15)
  const int mbase = m0 + ((lane >= 16) ? 8 : 0);
  const int ncol  = lane & 15;
  #pragma unroll
  for (int t = 0; t < NTILES; ++t) {
    int n = t * 16 + ncol;
    #pragma unroll
    for (int j = 0; j < 8; ++j) {
      int m = mbase + j;
      if (m < M && n < Nout) {
        float v = acc[t][j] + bias_s[n];
        if (relu) v = fmaxf(v, 0.0f);
        out[(size_t)m * ldo + n] = v;
      }
    }
  }
}

// ------------------------------------------------------- row log_softmax(40)
__global__ void gin_log_softmax40(float* __restrict__ out, int M) {
  int gw   = (blockIdx.x * blockDim.x + threadIdx.x) >> 5;
  int lane = threadIdx.x & 31;
  if (gw >= M) return;
  float* row = out + (size_t)gw * 40;
  float v0 = row[lane];
  float v1 = (lane < 8) ? row[32 + lane] : -3.4e38f;
  float mx = fmaxf(v0, v1);
  #pragma unroll
  for (int off = 16; off > 0; off >>= 1)
    mx = fmaxf(mx, __shfl_xor(mx, off, 32));
  float s = expf(v0 - mx) + ((lane < 8) ? expf(v1 - mx) : 0.0f);
  #pragma unroll
  for (int off = 16; off > 0; off >>= 1)
    s += __shfl_xor(s, off, 32);
  float lse = mx + logf(s);
  row[lane] = v0 - lse;
  if (lane < 8) row[32 + lane] = v1 - lse;
}

// ---------------------------------------------------------------- launcher
extern "C" void kernel_launch(void* const* d_in, const int* in_sizes, int n_in,
                              void* d_out, int out_size, void* d_ws, size_t ws_size,
                              hipStream_t stream) {
  const float*     x    = (const float*)d_in[0];
  const long long* ei   = (const long long*)d_in[1];
  const float*     w1a  = (const float*)d_in[2];
  const float*     b1a  = (const float*)d_in[3];
  const float*     w1b  = (const float*)d_in[4];
  const float*     b1b  = (const float*)d_in[5];
  const float*     eps1 = (const float*)d_in[6];
  const float*     w2a  = (const float*)d_in[7];
  const float*     b2a  = (const float*)d_in[8];
  const float*     w2b  = (const float*)d_in[9];
  const float*     b2b  = (const float*)d_in[10];
  const float*     eps2 = (const float*)d_in[11];

  const long long* src = ei;
  const long long* dst = ei + GIN_N_EDGES;

  float* ws   = (float*)d_ws;
  float* agg1 = ws;                                    // 50000*64
  float* hmid = agg1 + (size_t)GIN_N_NODES * 64;       // 50000*128
  float* h1   = hmid + (size_t)GIN_N_NODES * 128;      // 50000*128
  float* agg2 = h1   + (size_t)GIN_N_NODES * 128;      // 50000*128
  float* outp = (float*)d_out;                         // 50000*40

  const int mblocks = (GIN_N_NODES + 63) / 64;

  // ---- layer 1 ----
  {
    int n = GIN_N_NODES * 64;
    gin_zero_f32<<<(n + 255) / 256, 256, 0, stream>>>(agg1, n);
  }
  gin_scatter_add<6><<<((GIN_N_EDGES << 6) + 255) / 256, 256, 0, stream>>>(
      x, src, dst, agg1);
  gin_gemm_wmma<64, 8, true><<<mblocks, 128, 0, stream>>>(
      x, agg1, eps1, w1a, b1a, 128, hmid, 128, 128, GIN_N_NODES, 1);
  gin_gemm_wmma<128, 8, false><<<mblocks, 128, 0, stream>>>(
      hmid, nullptr, nullptr, w1b, b1b, 128, h1, 128, 128, GIN_N_NODES, 1);

  // ---- layer 2 ----
  {
    int n = GIN_N_NODES * 128;
    gin_zero_f32<<<(n + 255) / 256, 256, 0, stream>>>(agg2, n);
  }
  gin_scatter_add<7><<<((GIN_N_EDGES << 7) + 255) / 256, 256, 0, stream>>>(
      h1, src, dst, agg2);
  gin_gemm_wmma<128, 8, true><<<mblocks, 128, 0, stream>>>(
      h1, agg2, eps2, w2a, b2a, 128, hmid, 128, 128, GIN_N_NODES, 1);
  gin_gemm_wmma<128, 3, false><<<mblocks, 128, 0, stream>>>(
      hmid, nullptr, nullptr, w2b, b2b, 40, outp, 40, 40, GIN_N_NODES, 0);

  // ---- log_softmax ----
  {
    int blocks = (GIN_N_NODES * 32 + 127) / 128;
    gin_log_softmax40<<<blocks, 128, 0, stream>>>(outp, GIN_N_NODES);
  }
}
